// CoattentiveLayer_68624987456018
// MI455X (gfx1250) — matrix operand order repaired and checked by reading, not compiled, not run on hardware
//
#include <hip/hip_runtime.h>

// ---------------------------------------------------------------------------
// CoattentiveLayer for MI455X (gfx1250, wave32, WMMA bf16 16x16x32, f32 accum)
// B=16, Lc=2048, Lq=256, d=512. Padded: Lc+1 -> 2064, Lq+1 -> 272.
// ---------------------------------------------------------------------------

typedef __attribute__((ext_vector_type(16))) __bf16 v16bf;
typedef __attribute__((ext_vector_type(8)))  float v8f;

#define NB    16
#define LC    2048
#define LQ    256
#define DD    512
#define LCP   2064   // (Lc+1) padded to 16
#define LQP   272    // (Lq+1) padded to 16
#define LC1   2049   // valid context rows (incl. sentinel)
#define LQ1   257    // valid question rows (incl. sentinel)

union FragB16 {
    unsigned int   u32[8];
    v16bf v;
};

__device__ __forceinline__ unsigned short f2bf(float f) {
    unsigned int u = __builtin_bit_cast(unsigned int, f);
    unsigned int r = u + 0x7FFFu + ((u >> 16) & 1u);   // round-to-nearest-even
    return (unsigned short)(r >> 16);
}

// ---------------------------------------------------------------------------
// Pack fp32 inputs (+sentinel row, zero padding) into bf16 padded matrices
// ---------------------------------------------------------------------------
__global__ void pack_ctx_kernel(const float* __restrict__ context,
                                const float* __restrict__ sentinel,
                                unsigned short* __restrict__ dst) {
    long idx = (long)blockIdx.x * blockDim.x + threadIdx.x;
    long total = (long)NB * LCP * DD;
    if (idx >= total) return;
    int d = (int)(idx % DD);
    int r = (int)((idx / DD) % LCP);
    int b = (int)(idx / ((long)DD * LCP));
    float v = 0.0f;
    if (r == 0)              v = sentinel[d];
    else if (r <= LC)        v = context[((long)b * LC + (r - 1)) * DD + d];
    dst[idx] = f2bf(v);
}

__global__ void pack_q_kernel(const float* __restrict__ question,
                              const float* __restrict__ sentinel,
                              unsigned short* __restrict__ dst) {
    long idx = (long)blockIdx.x * blockDim.x + threadIdx.x;
    long total = (long)NB * LQP * DD;
    if (idx >= total) return;
    int d = (int)(idx % DD);
    int r = (int)((idx / DD) % LQP);
    int b = (int)(idx / ((long)DD * LQP));
    float v = 0.0f;
    if (r == 0)              v = sentinel[DD + d];
    else if (r <= LQ)        v = question[((long)b * LQ + (r - 1)) * DD + d];
    dst[idx] = f2bf(v);
}

// simple f32 -> bf16 conversion kernel (for packing proj_w)
__global__ void cvt_f32_bf16_kernel(const float* __restrict__ src,
                                    unsigned short* __restrict__ dst, long n) {
    long i = (long)blockIdx.x * blockDim.x + threadIdx.x;
    if (i < n) dst[i] = f2bf(src[i]);
}

// ---------------------------------------------------------------------------
// Generic bf16 GEMM, f32 accumulate: C[M,N] = A[M,K] * B[K,N]
//   A row-major bf16.  B: row-major (BTRANS=0) or B[k,n]=src[n*ldb+k] (BTRANS=1)
//   Tile 64x64, BK=32, 128 threads (4 waves, each wave a 32x32 subtile
//   of 2x2 v_wmma_f32_16x16x32_bf16 fragments).
// LDS layouts (stride 40 ushorts = 80 B, rows 16B-aligned):
//   As[row][k]  -> lane's A fragment = two contiguous 16B chunks (2x b128)
//   Bs[col][k]  -> lane's B fragment = two contiguous 16B chunks (2x b128)
// Epilogues:
//   EPI=0 : f32 store to outF (full)
//   EPI=1 : bf16 store of tanh(acc + bias[col]) to outB; rows>=validRows -> 0
//   EPI=2 : bf16 full store to outB + f32 store of rows [1,validRows) to
//           outF at (row-1)  (sum_of_* into strided d_out halves)
//   EPI=3 : f32 store of rows [1,validRows) to outF at (row-1)  (coattn_*)
// ---------------------------------------------------------------------------
template <int EPI, bool BTRANS>
__global__ __launch_bounds__(128) void gemm_bf16_kernel(
    const unsigned short* __restrict__ A, long sAb, int lda, int Mtot, int Ktot,
    const unsigned short* __restrict__ Bm, long sBb, int ldb, int Ntot,
    float* outF, long sFb, int ldF, int validRows,
    unsigned short* outB, long sBo, int ldBo,
    const float* __restrict__ bias)
{
    __shared__ __align__(16) unsigned short As[64 * 40];  // [row][k]
    __shared__ __align__(16) unsigned short Bs[64 * 40];  // [col][k]

    const int bz = blockIdx.z;
    A  += (long)bz * sAb;
    Bm += (long)bz * sBb;
    if (outF) outF += (long)bz * sFb;
    if (outB) outB += (long)bz * sBo;

    const int tid  = threadIdx.x;
    const int lane = tid & 31;
    const int w    = tid >> 5;
    const int wr   = w >> 1;          // wave row (0..1)
    const int wc   = w & 1;           // wave col (0..1)
    const int rowBase = blockIdx.y * 64;
    const int colBase = blockIdx.x * 64;
    const int half = lane >> 4;       // 0: lanes 0-15, 1: lanes 16-31
    const int ln   = lane & 15;

    const bool fullM = (rowBase + 64 <= Mtot);
    const bool fullN = (colBase + 64 <= Ntot);

    v8f acc[2][2] = {};

    const int nK = (Ktot + 31) >> 5;
    for (int kt = 0; kt < nK; ++kt) {
        const int k0 = kt * 32;
        const bool fullK = (k0 + 32 <= Ktot);

        // ---- stage A tile (64x32) into LDS [row][k], uint granularity ----
        if (fullM && fullK) {
            #pragma unroll
            for (int i = 0; i < 8; ++i) {
                int idx = tid + i * 128;            // 1024 uints
                int r = idx >> 4, cp = idx & 15;
                unsigned int val =
                    *(const unsigned int*)(A + (long)(rowBase + r) * lda + k0 + cp * 2);
                *(unsigned int*)(&As[r * 40 + cp * 2]) = val;
            }
        } else {
            #pragma unroll
            for (int i = 0; i < 8; ++i) {
                int idx = tid + i * 128;
                int r = idx >> 4, cp = idx & 15;
                int gr = rowBase + r, gk = k0 + cp * 2;
                unsigned int val = 0;
                if (gr < Mtot && gk < Ktot)
                    val = *(const unsigned int*)(A + (long)gr * lda + gk);
                *(unsigned int*)(&As[r * 40 + cp * 2]) = val;
            }
        }

        // ---- stage B tile (32x64) into LDS [col][k] ----
        if (BTRANS) {
            // src[n*ldb + k]: a uint read is a (k,k+1) pair for one column ->
            // stores directly into [col][k] layout, aligned b32.
            if (fullN && fullK) {
                #pragma unroll
                for (int i = 0; i < 8; ++i) {
                    int idx = tid + i * 128;        // 64 cols x 16 k-uints
                    int n = idx >> 4, kp = idx & 15;
                    unsigned int val =
                        *(const unsigned int*)(Bm + (long)(colBase + n) * ldb + k0 + kp * 2);
                    *(unsigned int*)(&Bs[n * 40 + kp * 2]) = val;
                }
            } else {
                #pragma unroll
                for (int i = 0; i < 8; ++i) {
                    int idx = tid + i * 128;
                    int n = idx >> 4, kp = idx & 15;
                    int gn = colBase + n, gk = k0 + kp * 2;
                    unsigned int val = 0;
                    if (gn < Ntot && gk < Ktot)
                        val = *(const unsigned int*)(Bm + (long)gn * ldb + gk);
                    *(unsigned int*)(&Bs[n * 40 + kp * 2]) = val;
                }
            }
        } else {
            // row-major src[k*ldb + n]: a uint read is (n,n+1) at one k ->
            // split into two b16 stores into [col][k] layout.
            if (fullN && fullK) {
                #pragma unroll
                for (int i = 0; i < 8; ++i) {
                    int idx = tid + i * 128;        // 32 k-rows x 32 col-uints
                    int r = idx >> 5, cp = idx & 31;
                    unsigned int val =
                        *(const unsigned int*)(Bm + (long)(k0 + r) * ldb + colBase + cp * 2);
                    Bs[(cp * 2 + 0) * 40 + r] = (unsigned short)(val & 0xFFFFu);
                    Bs[(cp * 2 + 1) * 40 + r] = (unsigned short)(val >> 16);
                }
            } else {
                #pragma unroll
                for (int i = 0; i < 8; ++i) {
                    int idx = tid + i * 128;
                    int r = idx >> 5, cp = idx & 31;
                    int gk = k0 + r, gn = colBase + cp * 2;
                    unsigned int val = 0;
                    if (gk < Ktot && gn < Ntot)
                        val = *(const unsigned int*)(Bm + (long)gk * ldb + gn);
                    Bs[(cp * 2 + 0) * 40 + r] = (unsigned short)(val & 0xFFFFu);
                    Bs[(cp * 2 + 1) * 40 + r] = (unsigned short)(val >> 16);
                }
            }
        }
        __syncthreads();

        // ---- assemble fragments (contiguous 16B chunks -> ds_load_b128) ----
        FragB16 afr[2], bfr[2];
        #pragma unroll
        for (int mi = 0; mi < 2; ++mi) {
            int row = wr * 32 + mi * 16 + ln;       // A: lane holds row M=ln
            const unsigned short* p = &As[row * 40 + half * 16];
            *(uint4*)(&afr[mi].u32[0]) = *(const uint4*)(p);
            *(uint4*)(&afr[mi].u32[4]) = *(const uint4*)(p + 8);
        }
        #pragma unroll
        for (int ni = 0; ni < 2; ++ni) {
            int col = wc * 32 + ni * 16 + ln;       // B: lane holds col N=ln
            const unsigned short* p = &Bs[col * 40 + half * 16];
            *(uint4*)(&bfr[ni].u32[0]) = *(const uint4*)(p);
            *(uint4*)(&bfr[ni].u32[4]) = *(const uint4*)(p + 8);
        }
        #pragma unroll
        for (int mi = 0; mi < 2; ++mi)
            #pragma unroll
            for (int ni = 0; ni < 2; ++ni)
                acc[mi][ni] = __builtin_amdgcn_wmma_f32_16x16x32_bf16(
                    false, afr[mi].v, false, bfr[ni].v,
                    (short)0, acc[mi][ni], false, false);
        __syncthreads();
    }

    // ---- epilogue (C/D layout: VGPR r -> M = r + half*8, N = ln) ----
    #pragma unroll
    for (int mi = 0; mi < 2; ++mi) {
        #pragma unroll
        for (int ni = 0; ni < 2; ++ni) {
            int fr = rowBase + wr * 32 + mi * 16 + half * 8;
            int fc = colBase + wc * 32 + ni * 16 + ln;
            #pragma unroll
            for (int r = 0; r < 8; ++r) {
                int grow = fr + r;
                int gcol = fc;
                float v = acc[mi][ni][r];
                if (EPI == 0) {
                    if (grow < Mtot && gcol < Ntot)
                        outF[(long)grow * ldF + gcol] = v;
                } else if (EPI == 1) {
                    if (grow < Mtot && gcol < Ntot) {
                        float t = (grow < validRows) ? tanhf(v + bias[gcol]) : 0.0f;
                        outB[(long)grow * ldBo + gcol] = f2bf(t);
                    }
                } else if (EPI == 2) {
                    if (grow < Mtot && gcol < Ntot) {
                        outB[(long)grow * ldBo + gcol] = f2bf(v);
                        if (grow >= 1 && grow < validRows)
                            outF[(long)(grow - 1) * ldF + gcol] = v;
                    }
                } else { // EPI == 3
                    if (gcol < Ntot && grow >= 1 && grow < validRows)
                        outF[(long)(grow - 1) * ldF + gcol] = v;
                }
            }
        }
    }
}

// ---------------------------------------------------------------------------
// Row softmax over q (masked by qpad) of affinity[b,c,:]; output AoqT bf16
// (same [c,q] layout).  One wave32 per row; __shfl_xor reductions.
// ---------------------------------------------------------------------------
__global__ __launch_bounds__(32) void row_softmax_kernel(
    const float* __restrict__ aff, const int* __restrict__ qpad,
    unsigned short* __restrict__ aoqT)
{
    const int c = blockIdx.x;
    const int b = blockIdx.y;
    const float* row = aff + ((long)b * LCP + c) * LQP;
    unsigned short* orow = aoqT + ((long)b * LCP + c) * LQP;
    const int* qp = qpad + (long)b * LQ;
    const int lane = threadIdx.x;

    float xs[9]; bool vs[9];
    float m = -3.0e38f;
    #pragma unroll
    for (int it = 0; it < 9; ++it) {
        int q = lane + it * 32;
        bool valid = false; float x = -3.0e38f;
        if (q < LQP) {
            valid = (q == 0) || (q < LQ1 && qp[q - 1] == 0);
            if (valid) x = row[q];
        }
        xs[it] = x; vs[it] = valid;
        m = fmaxf(m, x);
    }
    #pragma unroll
    for (int off = 16; off >= 1; off >>= 1)
        m = fmaxf(m, __shfl_xor(m, off, 32));
    float es[9]; float s = 0.0f;
    #pragma unroll
    for (int it = 0; it < 9; ++it) {
        float e = vs[it] ? expf(xs[it] - m) : 0.0f;
        es[it] = e; s += e;
    }
    #pragma unroll
    for (int off = 16; off >= 1; off >>= 1)
        s += __shfl_xor(s, off, 32);
    float inv = 1.0f / s;
    #pragma unroll
    for (int it = 0; it < 9; ++it) {
        int q = lane + it * 32;
        if (q < LQP) orow[q] = f2bf(es[it] * inv);
    }
}

// ---------------------------------------------------------------------------
// Column softmax over c (masked by cpad); output AocT bf16 in transposed
// layout [b, q, c].  One thread per (b, q) column; online max/sum then write.
// ---------------------------------------------------------------------------
__global__ __launch_bounds__(288) void col_softmax_kernel(
    const float* __restrict__ aff, const int* __restrict__ cpad,
    unsigned short* __restrict__ aocT)
{
    const int b = blockIdx.x;
    const int q = threadIdx.x;
    if (q >= LQP) return;
    const float* base = aff + (long)b * LCP * LQP + q;
    const int* cp = cpad + (long)b * LC;

    float m = -3.0e38f, s = 0.0f;
    for (int c = 0; c < LCP; ++c) {
        bool valid = (c == 0) || (c < LC1 && cp[c - 1] == 0);
        if (!valid) continue;
        float x = base[(long)c * LQP];
        if (x > m) { s = s * expf(m - x) + 1.0f; m = x; }
        else       { s += expf(x - m); }
    }
    float inv = 1.0f / s;
    unsigned short* out = aocT + ((long)b * LQP + q) * LCP;
    for (int c = 0; c < LCP; ++c) {
        bool valid = (c == 0) || (c < LC1 && cp[c - 1] == 0);
        float e = valid ? expf(base[(long)c * LQP] - m) * inv : 0.0f;
        out[c] = f2bf(e);
    }
}

// ---------------------------------------------------------------------------
// Launch
// ---------------------------------------------------------------------------
extern "C" void kernel_launch(void* const* d_in, const int* in_sizes, int n_in,
                              void* d_out, int out_size, void* d_ws, size_t ws_size,
                              hipStream_t stream) {
    (void)in_sizes; (void)n_in; (void)out_size; (void)ws_size;

    const float* context  = (const float*)d_in[0];   // [16,2048,512]
    const float* question = (const float*)d_in[1];   // [16,256,512]
    const int*   cpad     = (const int*)d_in[2];     // [16,2048]
    const int*   qpad     = (const int*)d_in[3];     // [16,256]
    const float* projW    = (const float*)d_in[4];   // [512,512]
    const float* projB    = (const float*)d_in[5];   // [512]
    const float* sentinel = (const float*)d_in[6];   // [2,512]

    float* out_c = (float*)d_out;                              // [16,2048,1024]
    float* out_q = out_c + (long)NB * LC * (2 * DD);           // [16,256,1024]

    // workspace carve-up (256B aligned)
    char* ws = (char*)d_ws;
    size_t off = 0;
    auto carve = [&](size_t bytes) {
        void* p = ws + off;
        off += (bytes + 255) & ~(size_t)255;
        return p;
    };
    unsigned short* ctxb = (unsigned short*)carve((size_t)NB * LCP * DD * 2);
    unsigned short* qinb = (unsigned short*)carve((size_t)NB * LQP * DD * 2);
    unsigned short* qstb = (unsigned short*)carve((size_t)NB * LQP * DD * 2);
    float*          aff  = (float*)         carve((size_t)NB * LCP * LQP * 4);
    unsigned short* aoqT = (unsigned short*)carve((size_t)NB * LCP * LQP * 2);
    unsigned short* aocT = (unsigned short*)carve((size_t)NB * LQP * LCP * 2);
    unsigned short* socb = (unsigned short*)carve((size_t)NB * LQP * DD * 2);
    unsigned short* soqb = (unsigned short*)carve((size_t)NB * LCP * DD * 2);
    unsigned short* wb   = (unsigned short*)carve((size_t)DD * DD * 2);

    // 1) pack bf16 padded inputs + weights
    {
        long t1 = (long)NB * LCP * DD;
        pack_ctx_kernel<<<dim3((unsigned)((t1 + 255) / 256)), 256, 0, stream>>>(
            context, sentinel, ctxb);
        long t2 = (long)NB * LQP * DD;
        pack_q_kernel<<<dim3((unsigned)((t2 + 255) / 256)), 256, 0, stream>>>(
            question, sentinel, qinb);
        long t3 = (long)DD * DD;
        cvt_f32_bf16_kernel<<<dim3((unsigned)((t3 + 255) / 256)), 256, 0, stream>>>(
            projW, wb, t3);
    }

    // 2) G1: qst = tanh(q_in @ W^T + b)   [per batch 272x512x512]
    gemm_bf16_kernel<1, true><<<dim3(8, 5, NB), 128, 0, stream>>>(
        qinb, (long)LQP * DD, DD, LQP, DD,
        wb, 0, DD, DD,
        nullptr, 0, 0, LQ1,
        qstb, (long)LQP * DD, DD,
        projB);

    // 3) G2: affinity[c,q] = ctx @ qst^T   -> f32
    gemm_bf16_kernel<0, true><<<dim3(5, 33, NB), 128, 0, stream>>>(
        ctxb, (long)LCP * DD, DD, LCP, DD,
        qstb, (long)LQP * DD, DD, LQP,
        aff, (long)LCP * LQP, LQP, 0,
        nullptr, 0, 0, nullptr);

    // 4) softmaxes -> bf16 attention operands
    row_softmax_kernel<<<dim3(LCP, NB), 32, 0, stream>>>(aff, qpad, aoqT);
    col_softmax_kernel<<<dim3(NB), 288, 0, stream>>>(aff, cpad, aocT);

    // 5) G4: sum_of_context[q,d] = AocT @ ctx  -> out_q[:, :, 512:] + bf16 ws
    gemm_bf16_kernel<2, false><<<dim3(8, 5, NB), 128, 0, stream>>>(
        aocT, (long)LQP * LCP, LCP, LQP, LCP,
        ctxb, (long)LCP * DD, DD, DD,
        out_q + DD, (long)LQ * (2 * DD), 2 * DD, LQ1,
        socb, (long)LQP * DD, DD,
        nullptr);

    // 6) G5: sum_of_question[c,d] = AoqT @ qst -> out_c[:, :, 512:] + bf16 ws
    gemm_bf16_kernel<2, false><<<dim3(8, 33, NB), 128, 0, stream>>>(
        aoqT, (long)LCP * LQP, LQP, LCP, LQP,
        qstb, (long)LQP * DD, DD, DD,
        out_c + DD, (long)LC * (2 * DD), 2 * DD, LC1,
        soqb, (long)LCP * DD, DD,
        nullptr);

    // 7) G6: coattn_context[c,d] = AoqT @ soc -> out_c[:, :, :512]
    gemm_bf16_kernel<3, false><<<dim3(8, 33, NB), 128, 0, stream>>>(
        aoqT, (long)LCP * LQP, LQP, LCP, LQP,
        socb, (long)LQP * DD, DD, DD,
        out_c, (long)LC * (2 * DD), 2 * DD, LC1,
        nullptr, 0, 0, nullptr);

    // 8) G7: coattn_question[q,d] = AocT @ soq -> out_q[:, :, :512]
    gemm_bf16_kernel<3, false><<<dim3(8, 5, NB), 128, 0, stream>>>(
        aocT, (long)LQP * LCP, LCP, LQP, LCP,
        soqb, (long)LCP * DD, DD, DD,
        out_q, (long)LQ * (2 * DD), 2 * DD, LQ1,
        nullptr, 0, 0, nullptr);
}
